// CausalSelfAttention_27994596836244
// MI455X (gfx1250) — matrix operand order repaired and checked by reading, not compiled, not run on hardware
//
#include <hip/hip_runtime.h>
#include <hip/hip_bf16.h>
#include <math.h>
#include <stdint.h>

typedef __attribute__((ext_vector_type(16))) __bf16 v16bf;
typedef __attribute__((ext_vector_type(8)))  float  v8f;
typedef unsigned int u32x4 __attribute__((ext_vector_type(4)));
typedef int          i32x8 __attribute__((ext_vector_type(8)));
typedef int          i32x4 __attribute__((ext_vector_type(4)));

#define BD 4
#define TD 2048
#define CD 1024
#define HN 16
#define DH 64
#define MROWS (BD*TD)   // 8192

// ---------- WMMA fragment loaders from LDS (bf16, per CDNA5 ISA layouts) ----------
__device__ __forceinline__ v16bf load_a_frag(const __bf16* base, int ld, int row, int kb, int lane) {
    const __bf16* p = base + row * ld + kb + ((lane & 16) ? 8 : 0);
    union { uint4 u[2]; v16bf v; } r;
    r.u[0] = *(const uint4*)(p);
    r.u[1] = *(const uint4*)(p + 16);
    return r.v;
}
__device__ __forceinline__ v16bf load_b_frag(const __bf16* base, int ld, int col, int kb, int lane) {
    const __bf16* p = base + col * ld + kb + ((lane & 16) ? 8 : 0);
    union { uint4 u[2]; v16bf v; } r;
    r.u[0] = *(const uint4*)(p);
    r.u[1] = *(const uint4*)(p + 16);
    return r.v;
}

#define WMMA_BF16(A, B, C) \
    __builtin_amdgcn_wmma_f32_16x16x32_bf16(false, (A), false, (B), (short)0, (C), false, false)

// ---------- Tensor Data Mover: 2D bf16 tile, global -> LDS ----------
// D# per CDNA5 ISA: group0 = {flags, lds_addr, global_addr, type=2},
// group1 = {data_size, tensor dims, tile dims, strides}.  6-arg clang-23 builtin form.
__device__ __forceinline__ void tdm_load_2d_bf16(
    uint32_t lds_addr, const void* gptr,
    uint32_t tile_cols, uint32_t tile_rows,
    uint32_t tensor_cols, uint32_t tensor_rows, uint32_t row_stride_elems)
{
    uint64_t ga = (uint64_t)(uintptr_t)gptr;
    u32x4 g0;
    g0[0] = 1u;                                                  // count=1, user mode, no gather
    g0[1] = lds_addr;                                            // LDS byte address
    g0[2] = (uint32_t)ga;                                        // global_addr[31:0]
    g0[3] = (uint32_t)((ga >> 32) & 0x01FFFFFFu) | (2u << 30);   // global_addr[56:32], type=2
    i32x8 g1;
    g1[0] = (int)(1u << 16);                                     // data_size=1 (2 bytes)
    g1[1] = (int)((tensor_cols & 0xFFFFu) << 16);                // tensor_dim0 lo16
    g1[2] = (int)(((tensor_cols >> 16) & 0xFFFFu) |
                  ((tensor_rows & 0xFFFFu) << 16));              // tensor_dim0 hi / tensor_dim1 lo
    g1[3] = (int)(((tensor_rows >> 16) & 0xFFFFu) |
                  (tile_cols << 16));                            // tensor_dim1 hi / tile_dim0
    g1[4] = (int)tile_rows;                                      // tile_dim1 (tile_dim2=0)
    g1[5] = (int)row_stride_elems;                               // tensor_dim0_stride lo32
    g1[6] = 0;                                                   // stride hi / dim1_stride
    g1[7] = 0;
    i32x4 z4 = {0, 0, 0, 0};
    i32x8 z8 = {0, 0, 0, 0, 0, 0, 0, 0};
    __builtin_amdgcn_tensor_load_to_lds(g0, g1, z4, z4, z8, 0);
}
__device__ __forceinline__ uint32_t lds_addr_of(const void* p) {
    // generic LDS pointer: addr[31:0] is the LDS byte offset (flat->LDS truncation)
    return (uint32_t)(uintptr_t)p;
}

// =====================================================================
// Kernel 1: QKV GEMM  (x[8192x1024] @ w[1024x3072] + b) -> bf16 q/k (B,H,T,hd),
// v transposed (B,H,hd,T).  128x128 tile, 256 threads = 8 waves (2x4).
// =====================================================================
__global__ __launch_bounds__(256) void qkv_gemm(
    const float* __restrict__ x, const float* __restrict__ w, const float* __restrict__ bias,
    __bf16* __restrict__ qws, __bf16* __restrict__ kws, __bf16* __restrict__ vtws)
{
    const int N = 3 * CD, K = CD;
    __shared__ __align__(16) __bf16 sA[128 * 32];   // [m][k]
    __shared__ __align__(16) __bf16 sB[128 * 32];   // [n][k]

    const int tid  = threadIdx.x;
    const int lane = tid & 31;
    const int wave = tid >> 5;
    const int wm   = wave >> 2;
    const int wn   = wave & 3;
    const int m0   = blockIdx.y * 128;
    const int n0   = blockIdx.x * 128;

    v8f acc[4][2];
#pragma unroll
    for (int i = 0; i < 4; ++i)
#pragma unroll
        for (int j = 0; j < 2; ++j) acc[i][j] = {};

    const int arow = tid >> 1, aseg = (tid & 1) * 16;
    const int bk   = tid >> 3, bseg = (tid & 7) * 16;

    for (int k0 = 0; k0 < K; k0 += 32) {
        const float* gA = x + (size_t)(m0 + arow) * K + k0 + aseg;
#pragma unroll
        for (int i = 0; i < 16; i += 4) {
            float4 f = *(const float4*)(gA + i);
            __bf16* d = &sA[arow * 32 + aseg + i];
            d[0] = (__bf16)f.x; d[1] = (__bf16)f.y; d[2] = (__bf16)f.z; d[3] = (__bf16)f.w;
        }
        const float* gB = w + (size_t)(k0 + bk) * N + n0 + bseg;
#pragma unroll
        for (int i = 0; i < 16; i += 4) {
            float4 f = *(const float4*)(gB + i);
            sB[(bseg + i + 0) * 32 + bk] = (__bf16)f.x;
            sB[(bseg + i + 1) * 32 + bk] = (__bf16)f.y;
            sB[(bseg + i + 2) * 32 + bk] = (__bf16)f.z;
            sB[(bseg + i + 3) * 32 + bk] = (__bf16)f.w;
        }
        if (k0 + 32 < K) {
            __builtin_prefetch(gA + 32, 0, 0);
            __builtin_prefetch(w + (size_t)(k0 + 32 + bk) * N + n0 + bseg, 0, 0);
        }
        __syncthreads();

        v16bf afr[4];
#pragma unroll
        for (int mt = 0; mt < 4; ++mt)
            afr[mt] = load_a_frag(sA, 32, wm * 64 + mt * 16 + (lane & 15), 0, lane);
#pragma unroll
        for (int nt = 0; nt < 2; ++nt) {
            v16bf bfr = load_b_frag(sB, 32, wn * 32 + nt * 16 + (lane & 15), 0, lane);
#pragma unroll
            for (int mt = 0; mt < 4; ++mt)
                acc[mt][nt] = WMMA_BF16(afr[mt], bfr, acc[mt][nt]);
        }
        __syncthreads();
    }

    const int mbase = (lane & 16) ? 8 : 0;
#pragma unroll
    for (int mt = 0; mt < 4; ++mt)
#pragma unroll
        for (int nt = 0; nt < 2; ++nt)
#pragma unroll
            for (int r = 0; r < 8; ++r) {
                int gm = m0 + wm * 64 + mt * 16 + mbase + r;     // b*T + t
                int gn = n0 + wn * 32 + nt * 16 + (lane & 15);   // 0..3071
                float v = acc[mt][nt][r] + bias[gn];
                int b = gm >> 11, t = gm & (TD - 1);
                int which = gn >> 10, nn = gn & (CD - 1);
                int h = nn >> 6, d = nn & 63;
                size_t bh = (size_t)(b * HN + h);
                if (which == 0)      qws[(bh * TD + t) * DH + d] = (__bf16)v;
                else if (which == 1) kws[(bh * TD + t) * DH + d] = (__bf16)v;
                else                 vtws[(bh * DH + d) * TD + t] = (__bf16)v;   // transposed
            }
}

// =====================================================================
// Kernel 2: RoPE on bf16 q,k in-place (layout (B,H,T,hd))
// =====================================================================
__global__ __launch_bounds__(256) void rope_kernel(__bf16* __restrict__ qws, __bf16* __restrict__ kws)
{
    size_t i = (size_t)blockIdx.x * blockDim.x + threadIdx.x;   // over B*H*T*32
    if (i >= (size_t)BD * HN * TD * 32) return;
    int d = (int)(i & 31);
    int t = (int)((i >> 5) & (TD - 1));
    size_t bh = i >> 16;
    size_t off = (bh * TD + t) * DH;
    float freq = __expf(-((float)(2 * d) / (float)DH) * logf(10000.0f));
    float ang = (float)t * freq;
    float c = __cosf(ang), s = __sinf(ang);
    float q1 = (float)qws[off + d], q2 = (float)qws[off + d + 32];
    qws[off + d]      = (__bf16)(q1 * c - q2 * s);
    qws[off + d + 32] = (__bf16)(q1 * s + q2 * c);
    float k1 = (float)kws[off + d], k2 = (float)kws[off + d + 32];
    kws[off + d]      = (__bf16)(k1 * c - k2 * s);
    kws[off + d + 32] = (__bf16)(k1 * s + k2 * c);
}

// =====================================================================
// Kernel 3: flash attention, causal.  64 q-rows per block, 128 threads = 4 waves.
// TDM stages Q/K/V tiles straight into WMMA-fragment LDS layouts (V pre-transposed
// in the workspace so no LDS transpose is needed).  bf16 WMMA, f32 accumulate.
// =====================================================================
__global__ __launch_bounds__(128) void flash_attn(
    const __bf16* __restrict__ qws, const __bf16* __restrict__ kws,
    const __bf16* __restrict__ vtws, __bf16* __restrict__ ctx)
{
    __shared__ __align__(16) __bf16 sQ [64 * 64];   // [row][d]
    __shared__ __align__(16) __bf16 sK [64 * 64];   // [key][d]  (B-frag: col=key)
    __shared__ __align__(16) __bf16 sVT[64 * 64];   // [d][key]  (B-frag: col=d)
    __shared__ __align__(16) float  sS [64 * 64];
    __shared__ __align__(16) __bf16 sP [64 * 64];
    __shared__ float sFac[64];

    const int tid  = threadIdx.x;
    const int lane = tid & 31;
    const int wave = tid >> 5;
    const int mb = blockIdx.x, bh = blockIdx.y;
    const int t0 = mb * 64;
    const int rb = wave * 16;
    const int mbase = (lane & 16) ? 8 : 0;
    const size_t base = (size_t)bh * TD * DH;    // (B,H,T,hd) / (B,H,hd,T) both bh*T*64

    // stage Q via TDM (wave 0 issues; descriptor is block-uniform)
    if (tid < 32) {
        tdm_load_2d_bf16(lds_addr_of(sQ), qws + base + (size_t)t0 * DH,
                         /*tile*/ DH, 64, /*tensor*/ DH, TD, /*stride*/ DH);
        __builtin_amdgcn_s_wait_tensorcnt(0);
    }
    __syncthreads();

    v16bf qa0 = load_a_frag(sQ, 64, rb + (lane & 15), 0, lane);
    v16bf qa1 = load_a_frag(sQ, 64, rb + (lane & 15), 32, lane);

    v8f oacc[4];
#pragma unroll
    for (int nt = 0; nt < 4; ++nt) oacc[nt] = {};
    float m_i = -INFINITY, l_i = 0.0f;   // threads 0..63 own row = tid

    for (int kb = 0; kb <= mb; ++kb) {
        const int k0 = kb * 64;
        if (tid < 32) {
            // K tile: 64 keys x 64 d, rows contiguous (stride DH)
            tdm_load_2d_bf16(lds_addr_of(sK), kws + base + (size_t)k0 * DH,
                             DH, 64, DH, TD, DH);
            // V^T tile: 64 d-rows x 64 keys from (B,H,hd,T), row stride T
            tdm_load_2d_bf16(lds_addr_of(sVT), vtws + base + k0,
                             64, DH, TD, DH, TD);
            __builtin_amdgcn_s_wait_tensorcnt(0);
        }
        __syncthreads();

        // S = Q K^T * scale, causal mask -> sS
#pragma unroll
        for (int nt = 0; nt < 4; ++nt) {
            v8f s = {};
            v16bf b0 = load_b_frag(sK, 64, nt * 16 + (lane & 15), 0, lane);
            v16bf b1 = load_b_frag(sK, 64, nt * 16 + (lane & 15), 32, lane);
            s = WMMA_BF16(qa0, b0, s);
            s = WMMA_BF16(qa1, b1, s);
            int col  = nt * 16 + (lane & 15);
            int colg = k0 + col;
#pragma unroll
            for (int r = 0; r < 8; ++r) {
                int rowl = rb + mbase + r;
                float v = s[r] * 0.125f;                 // 1/sqrt(64)
                if (colg > t0 + rowl) v = -INFINITY;
                sS[rowl * 64 + col] = v;
            }
        }
        __syncthreads();

        // online softmax per row
        if (tid < 64) {
            const float* srow = &sS[tid * 64];
            float mx = m_i;
#pragma unroll 8
            for (int j = 0; j < 64; ++j) mx = fmaxf(mx, srow[j]);
            float fac = __expf(m_i - mx);
            float sum = 0.0f;
#pragma unroll 8
            for (int j = 0; j < 64; ++j) {
                float p = __expf(srow[j] - mx);
                sP[tid * 64 + j] = (__bf16)p;
                sum += p;
            }
            l_i = l_i * fac + sum;
            m_i = mx;
            sFac[tid] = fac;
        }
        __syncthreads();

        // O = O*fac + P @ V
        float f[8];
#pragma unroll
        for (int r = 0; r < 8; ++r) f[r] = sFac[rb + mbase + r];
        v16bf pa0 = load_a_frag(sP, 64, rb + (lane & 15), 0, lane);
        v16bf pa1 = load_a_frag(sP, 64, rb + (lane & 15), 32, lane);
#pragma unroll
        for (int nt = 0; nt < 4; ++nt) {
#pragma unroll
            for (int r = 0; r < 8; ++r) oacc[nt][r] *= f[r];
            v16bf b0 = load_b_frag(sVT, 64, nt * 16 + (lane & 15), 0, lane);
            v16bf b1 = load_b_frag(sVT, 64, nt * 16 + (lane & 15), 32, lane);
            oacc[nt] = WMMA_BF16(pa0, b0, oacc[nt]);
            oacc[nt] = WMMA_BF16(pa1, b1, oacc[nt]);
        }
        __syncthreads();
    }

    if (tid < 64) sFac[tid] = 1.0f / l_i;
    __syncthreads();

    float inv[8];
#pragma unroll
    for (int r = 0; r < 8; ++r) inv[r] = sFac[rb + mbase + r];
    int b = bh >> 4, h = bh & (HN - 1);
#pragma unroll
    for (int nt = 0; nt < 4; ++nt)
#pragma unroll
        for (int r = 0; r < 8; ++r) {
            int t = t0 + rb + mbase + r;
            int d = nt * 16 + (lane & 15);
            ctx[((size_t)(b * TD + t)) * CD + h * DH + d] = (__bf16)(oacc[nt][r] * inv[r]);
        }
}

// =====================================================================
// Kernel 4: output projection  ctx(bf16)[8192x1024] @ w_proj[1024x1024] + b -> out(f32)
// A tile staged by TDM (bf16, no conversion); B staged with cvt+transpose.
// =====================================================================
__global__ __launch_bounds__(256) void proj_gemm(
    const __bf16* __restrict__ a, const float* __restrict__ w, const float* __restrict__ bias,
    float* __restrict__ out)
{
    const int N = CD, K = CD;
    __shared__ __align__(16) __bf16 sA[128 * 32];   // [m][k] via TDM
    __shared__ __align__(16) __bf16 sB[128 * 32];   // [n][k]

    const int tid  = threadIdx.x;
    const int lane = tid & 31;
    const int wave = tid >> 5;
    const int wm   = wave >> 2;
    const int wn   = wave & 3;
    const int m0   = blockIdx.y * 128;
    const int n0   = blockIdx.x * 128;

    v8f acc[4][2];
#pragma unroll
    for (int i = 0; i < 4; ++i)
#pragma unroll
        for (int j = 0; j < 2; ++j) acc[i][j] = {};

    const int bk = tid >> 3, bseg = (tid & 7) * 16;

    for (int k0 = 0; k0 < K; k0 += 32) {
        if (tid < 32) {  // TDM: 128 rows x 32 bf16 cols, tensor (8192 x 1024)
            tdm_load_2d_bf16(lds_addr_of(sA), a + (size_t)m0 * K + k0,
                             /*tile*/ 32, 128, /*tensor*/ K, MROWS, /*stride*/ K);
        }
        const float* gB = w + (size_t)(k0 + bk) * N + n0 + bseg;
#pragma unroll
        for (int i = 0; i < 16; i += 4) {
            float4 f = *(const float4*)(gB + i);
            sB[(bseg + i + 0) * 32 + bk] = (__bf16)f.x;
            sB[(bseg + i + 1) * 32 + bk] = (__bf16)f.y;
            sB[(bseg + i + 2) * 32 + bk] = (__bf16)f.z;
            sB[(bseg + i + 3) * 32 + bk] = (__bf16)f.w;
        }
        if (k0 + 32 < K)
            __builtin_prefetch(w + (size_t)(k0 + 32 + bk) * N + n0 + bseg, 0, 0);
        if (tid < 32) __builtin_amdgcn_s_wait_tensorcnt(0);
        __syncthreads();

        v16bf afr[4];
#pragma unroll
        for (int mt = 0; mt < 4; ++mt)
            afr[mt] = load_a_frag(sA, 32, wm * 64 + mt * 16 + (lane & 15), 0, lane);
#pragma unroll
        for (int nt = 0; nt < 2; ++nt) {
            v16bf bfr = load_b_frag(sB, 32, wn * 32 + nt * 16 + (lane & 15), 0, lane);
#pragma unroll
            for (int mt = 0; mt < 4; ++mt)
                acc[mt][nt] = WMMA_BF16(afr[mt], bfr, acc[mt][nt]);
        }
        __syncthreads();
    }

    const int mbase = (lane & 16) ? 8 : 0;
#pragma unroll
    for (int mt = 0; mt < 4; ++mt)
#pragma unroll
        for (int nt = 0; nt < 2; ++nt)
#pragma unroll
            for (int r = 0; r < 8; ++r) {
                int gm = m0 + wm * 64 + mt * 16 + mbase + r;
                int gn = n0 + wn * 32 + nt * 16 + (lane & 15);
                out[(size_t)gm * N + gn] = acc[mt][nt][r] + bias[gn];
            }
}

// =====================================================================
extern "C" void kernel_launch(void* const* d_in, const int* in_sizes, int n_in,
                              void* d_out, int out_size, void* d_ws, size_t ws_size,
                              hipStream_t stream)
{
    (void)in_sizes; (void)n_in; (void)out_size; (void)ws_size;
    const float* x      = (const float*)d_in[0];
    const float* w_qkv  = (const float*)d_in[1];
    const float* b_qkv  = (const float*)d_in[2];
    const float* w_proj = (const float*)d_in[3];
    const float* b_proj = (const float*)d_in[4];
    float* out = (float*)d_out;

    const size_t per = (size_t)BD * HN * TD * DH;   // 8.4M elements
    __bf16* qws  = (__bf16*)d_ws;
    __bf16* kws  = qws + per;
    __bf16* vtws = kws + per;                        // (B,H,hd,T)
    __bf16* ctx  = vtws + per;                       // (B,T,C)

    qkv_gemm<<<dim3(3 * CD / 128, MROWS / 128), 256, 0, stream>>>(x, w_qkv, b_qkv, qws, kws, vtws);
    {
        size_t total = (size_t)BD * HN * TD * 32;
        rope_kernel<<<(unsigned)((total + 255) / 256), 256, 0, stream>>>(qws, kws);
    }
    flash_attn<<<dim3(TD / 64, BD * HN), 128, 0, stream>>>(qws, kws, vtws, ctx);
    proj_gemm<<<dim3(CD / 128, MROWS / 128), 256, 0, stream>>>(ctx, w_proj, b_proj, out);
}
